// BPTAttentionWrapperWithAlibi_3487513444835
// MI455X (gfx1250) — matrix operand order repaired and checked by reading, not compiled, not run on hardware
//
#include <hip/hip_runtime.h>
#include <hip/hip_bf16.h>

// Problem constants (match reference)
#define NHEADS 16
#define HD     128
#define HID    2048
#define NQKV   6144          // 3*HID
#define BB     2
#define SS     2048
#define TT     (BB*SS)       // 4096 tokens
#define INV_NORM 0.08838834764831845f   // 1/sqrt(128)

typedef __attribute__((ext_vector_type(16))) __bf16 v16bf;
typedef __attribute__((ext_vector_type(8)))  float  v8f;

// float -> bf16 (round-to-nearest-even) without relying on cast codegen
static __device__ __forceinline__ __bf16 f2bf(float f) {
    union { float f; unsigned u; } a; a.f = f;
    unsigned r = a.u + 0x7FFFu + ((a.u >> 16) & 1u);
    union { unsigned short s; __bf16 b; } o; o.s = (unsigned short)(r >> 16);
    return o.b;
}

// CDNA5 async DMA: copy 16B/lane from global to LDS, tracked by ASYNCcnt.
// lds_off: wave-relative LDS byte address (low 32 bits of a generic LDS ptr).
static __device__ __forceinline__ void async_load_b128(unsigned lds_off, const void* gaddr) {
    asm volatile("global_load_async_to_lds_b128 %0, %1, off"
                 :: "v"(lds_off), "v"((unsigned long long)(uintptr_t)gaddr)
                 : "memory");
}

// ---------------------------------------------------------------------------
// Kernel 1: f32 -> bf16 cast, 4 elements/thread
// ---------------------------------------------------------------------------
__global__ __launch_bounds__(256)
void cast4_kernel(const float4* __restrict__ in, __bf16* __restrict__ out, long n4) {
    long i = (long)blockIdx.x * 256 + threadIdx.x;
    if (i >= n4) return;
    float4 v = in[i];
    union { __bf16 h[4]; unsigned long long u; } o;
    o.h[0] = f2bf(v.x); o.h[1] = f2bf(v.y); o.h[2] = f2bf(v.z); o.h[3] = f2bf(v.w);
    *(unsigned long long*)(out + i * 4) = o.u;
}

// ---------------------------------------------------------------------------
// Kernel 2/7: bf16 WMMA GEMM: C[M,N] = A[M,K] @ W[N,K]^T + bias (+residual)
// Block = 8 waves (2x4) -> 64x128 tile; wave = 32x32 (2x2 frags).
// K panels of width 64 are staged to LDS with global_load_async_to_lds_b128
// (double-buffered, ASYNCcnt-tracked), shared by all waves: L2 traffic drops
// ~2.6x vs per-wave global loads. Rows padded to 144B to spread LDS banks.
// Optional epilogue: extract q columns (j==0 of [H][3][D]) as bf16.
// ---------------------------------------------------------------------------
#define APAD_B   144                 // 64 K-elems * 2B = 128B, padded to 144B
#define ABUF_B   (64  * APAD_B)      //  9216 B
#define BBUF_B   (128 * APAD_B)      // 18432 B
#define BUF_B    (ABUF_B + BBUF_B)   // 27648 B per stage buffer

__global__ __launch_bounds__(256)
void wmma_gemm_kernel(const __bf16* __restrict__ A, const __bf16* __restrict__ W,
                      const float* __restrict__ bias, const float* __restrict__ residual,
                      float* __restrict__ C, __bf16* __restrict__ q_out,
                      int M, int N, int K) {
    __shared__ __align__(16) char smem[2 * BUF_B];   // 54 KB
    const int tid  = threadIdx.x;
    const int lane = tid & 31;
    const int wave = tid >> 5;
    const int wm   = wave >> 2;         // 0..1
    const int wn   = wave & 3;          // 0..3
    const int sel  = lane >> 4;
    const int r    = lane & 15;
    const int mblk = blockIdx.y * 64;
    const int nblk = blockIdx.x * 128;

    const __bf16* Apanel = A + (size_t)mblk * K;
    const __bf16* Wpanel = W + (size_t)nblk * K;
    const unsigned ldsbase = (unsigned)(uintptr_t)smem;

    // Stage one 64-wide K panel: A 64x64 (512 x 16B chunks, 2/thread),
    // B 128x64 (1024 chunks, 4/thread).
    auto issue = [&](int buf, int k0) {
#pragma unroll
        for (int i = 0; i < 2; ++i) {
            int c = tid + 256 * i;
            int row = c >> 3, chk = c & 7;
            async_load_b128(ldsbase + buf * BUF_B + row * APAD_B + chk * 16,
                            Apanel + (size_t)row * K + k0 + chk * 8);
        }
#pragma unroll
        for (int i = 0; i < 4; ++i) {
            int c = tid + 256 * i;
            int row = c >> 3, chk = c & 7;
            async_load_b128(ldsbase + buf * BUF_B + ABUF_B + row * APAD_B + chk * 16,
                            Wpanel + (size_t)row * K + k0 + chk * 8);
        }
    };

    v8f zero = {0.f,0.f,0.f,0.f,0.f,0.f,0.f,0.f};
    v8f acc[2][2] = {{zero, zero}, {zero, zero}};

    const int nsteps = K >> 6;          // K / 64
    issue(0, 0);
    for (int s = 0; s < nsteps; ++s) {
        const int k0 = s << 6;
        const int knext = (s + 1 < nsteps) ? (k0 + 64) : k0;  // tail: harmless reload
        issue((s + 1) & 1, knext);
        __builtin_prefetch(Apanel + (size_t)(tid >> 2) * K + k0 + 192, 0, 1);
        __builtin_prefetch(Wpanel + (size_t)(tid >> 1) * K + k0 + 192, 0, 1);
        asm volatile("s_wait_asynccnt 0x6" ::: "memory");  // this wave's panel done
        __syncthreads();                                   // all waves' parts done

        const char* bufp = smem + (s & 1) * BUF_B;
        const __bf16* Ab = (const __bf16*)bufp;            // row stride 72 elems
        const __bf16* Bb = (const __bf16*)(bufp + ABUF_B);
        union U { v16bf v; uint4 u[2]; };
#pragma unroll
        for (int ks = 0; ks < 64; ks += 32) {
            U a[2], b[2];
#pragma unroll
            for (int mi = 0; mi < 2; ++mi) {
                const __bf16* pa = Ab + (wm * 32 + mi * 16 + r) * 72 + ks;
                a[mi].u[0] = *(const uint4*)(pa + 16 * sel);
                a[mi].u[1] = *(const uint4*)(pa + 8 + 16 * sel);
            }
#pragma unroll
            for (int ni = 0; ni < 2; ++ni) {
                const __bf16* pb = Bb + (wn * 32 + ni * 16 + r) * 72 + ks + 16 * sel;
                b[ni].u[0] = ((const uint4*)pb)[0];
                b[ni].u[1] = ((const uint4*)pb)[1];
            }
            acc[0][0] = __builtin_amdgcn_wmma_f32_16x16x32_bf16(false, a[0].v, false, b[0].v, (short)0, acc[0][0], false, false);
            acc[0][1] = __builtin_amdgcn_wmma_f32_16x16x32_bf16(false, a[0].v, false, b[1].v, (short)0, acc[0][1], false, false);
            acc[1][0] = __builtin_amdgcn_wmma_f32_16x16x32_bf16(false, a[1].v, false, b[0].v, (short)0, acc[1][0], false, false);
            acc[1][1] = __builtin_amdgcn_wmma_f32_16x16x32_bf16(false, a[1].v, false, b[1].v, (short)0, acc[1][1], false, false);
        }
        __syncthreads();                                   // done with buf before overwrite
    }

    // Epilogue: C layout per VGPR v: lanes 0-15 -> M=v, lanes 16-31 -> M=8+v; N=r
    for (int mi = 0; mi < 2; ++mi) {
        for (int ni = 0; ni < 2; ++ni) {
            int rbase = mblk + wm * 32 + mi * 16 + sel * 8;
            int col   = nblk + wn * 32 + ni * 16 + r;
            float bcol = bias[col];
            for (int v = 0; v < 8; ++v) {
                int row = rbase + v;
                float val = acc[mi][ni][v] + bcol;
                if (residual) val += residual[(size_t)row * N + col];
                C[(size_t)row * N + col] = val;
                if (q_out) {
                    int h = col / 384;
                    int loc = col - h * 384;        // j*128 + d
                    if (loc < 128)                  // j == 0 -> q
                        q_out[(size_t)row * HID + h * 128 + loc] = f2bf(val);
                }
            }
        }
    }
}

// ---------------------------------------------------------------------------
// Kernel 3: Gram matrix G[bh] = q^T q (128x128, K=2048), one block per (b,h)
// ---------------------------------------------------------------------------
__global__ __launch_bounds__(256)
void gram_kernel(const float* __restrict__ fused, float* __restrict__ G) {
    __shared__ float qs[32 * 128];
    int bh = blockIdx.x;
    int b = bh >> 4, h = bh & 15;
    int tid = threadIdx.x;
    int tx = tid & 15, ty = tid >> 4;
    const float* base = fused + (size_t)b * SS * NQKV + h * 384;  // q component

    float acc[8][8];
    for (int i = 0; i < 8; ++i)
        for (int j = 0; j < 8; ++j) acc[i][j] = 0.f;

    for (int s0 = 0; s0 < SS; s0 += 32) {
        for (int i = 0; i < 16; ++i) {
            int idx = tid + i * 256;
            int sr = idx >> 7, d = idx & 127;
            qs[idx] = base[(size_t)(s0 + sr) * NQKV + d];
        }
        __syncthreads();
        for (int ss = 0; ss < 32; ++ss) {
            float arow[8], brow[8];
            for (int i = 0; i < 8; ++i) arow[i] = qs[ss * 128 + ty * 8 + i];
            for (int j = 0; j < 8; ++j) brow[j] = qs[ss * 128 + tx * 8 + j];
            for (int i = 0; i < 8; ++i)
                for (int j = 0; j < 8; ++j) acc[i][j] += arow[i] * brow[j];
        }
        __syncthreads();
    }
    float* Gb = G + (size_t)bh * HD * HD;
    for (int i = 0; i < 8; ++i)
        for (int j = 0; j < 8; ++j)
            Gb[(size_t)(ty * 8 + i) * HD + tx * 8 + j] = acc[i][j];
}

// ---------------------------------------------------------------------------
// Kernel 4: in-place Gauss-Jordan inverse of 128x128 SPD G (LDS-resident)
// ---------------------------------------------------------------------------
__global__ __launch_bounds__(256)
void invert128_kernel(float* __restrict__ G) {
    __shared__ float Ash[HD * HD];   // 64 KB
    int tid = threadIdx.x;
    float* Gb = G + (size_t)blockIdx.x * HD * HD;
    for (int i = 0; i < 64; ++i) Ash[tid + i * 256] = Gb[tid + i * 256];
    __syncthreads();

    const int irow  = tid >> 1;
    const int jbase = (tid & 1) * 64;
    for (int k = 0; k < HD; ++k) {
        float p = Ash[k * HD + k];
        __syncthreads();
        if (tid < HD) {
            float val = Ash[k * HD + tid];
            Ash[k * HD + tid] = (tid == k) ? (1.0f / p) : (val / p);
        }
        __syncthreads();
        float f = Ash[irow * HD + k];
        __syncthreads();
        if (irow != k) {
            float invp = Ash[k * HD + k];
            for (int jj = 0; jj < 64; ++jj) {
                int j = jbase + jj;
                float kj  = Ash[k * HD + j];
                float val = Ash[irow * HD + j];
                Ash[irow * HD + j] = (j == k) ? (-f * invp) : (val - f * kj);
            }
        }
        __syncthreads();
    }
    for (int i = 0; i < 64; ++i) Gb[tid + i * 256] = Ash[tid + i * 256];
}

// ---------------------------------------------------------------------------
// Kernel 5: k_off[b,s,h,:] = inv_norm*k + BETA*alibi[b,h,s] * (q_row @ Ginv)
// Ginv symmetric -> read rows of Ginv, coalesced across d. One block/(t,h).
// ---------------------------------------------------------------------------
__global__ __launch_bounds__(128)
void koff_kernel(const float* __restrict__ fused, const float* __restrict__ Ginv,
                 const float* __restrict__ alibi, __bf16* __restrict__ koffb) {
    __shared__ float qrow[HD];
    int blk = blockIdx.x;
    int t = blk >> 4, h = blk & 15;
    int b = t >> 11, s = t & 2047;
    int d = threadIdx.x;
    const float* frow = fused + (size_t)t * NQKV + h * 384;
    qrow[d] = frow[d];
    __syncthreads();
    const float* Gb = Ginv + (size_t)(b * NHEADS + h) * HD * HD;
    float dot = 0.f;
    for (int e = 0; e < HD; ++e) dot += qrow[e] * Gb[(size_t)e * HD + d];
    float al = alibi[(size_t)(b * NHEADS + h) * SS + s];
    float val = INV_NORM * frow[128 + d] + al * dot;   // BETA = 1.0
    koffb[(size_t)t * HID + h * 128 + d] = f2bf(val);
}

// ---------------------------------------------------------------------------
// Kernel 6: per-token head attention. One wave per token.
// scores(16x16,K=128) via 4x v_wmma_f32_16x16x32_bf16; softmax across columns
// (16-lane groups); ctx = attn @ v via LDS-staged attn + VALU FMAs.
// ---------------------------------------------------------------------------
__global__ __launch_bounds__(256)
void attn_kernel(const __bf16* __restrict__ qb, const __bf16* __restrict__ koffb,
                 const float* __restrict__ fused, __bf16* __restrict__ ctxb) {
    __shared__ float attn_lds[8][16][16];
    const int lane = threadIdx.x & 31;
    const int wave = threadIdx.x >> 5;
    const int t = blockIdx.x * 8 + wave;
    const int sel = lane >> 4, r = lane & 15;

    const __bf16* qrow = qb    + (size_t)t * HID;   // [head][d] row-major
    const __bf16* krow = koffb + (size_t)t * HID;

    v8f acc = {0.f,0.f,0.f,0.f,0.f,0.f,0.f,0.f};
    union U { v16bf v; uint4 u[2]; };
    for (int k0 = 0; k0 < HD; k0 += 32) {
        U a, bfr;
        const __bf16* pa = qrow + r * HD + k0;
        a.u[0] = *(const uint4*)(pa +     16 * sel);
        a.u[1] = *(const uint4*)(pa + 8 + 16 * sel);
        const __bf16* pb = krow + r * HD + k0 + 16 * sel;
        bfr.u[0] = ((const uint4*)pb)[0];
        bfr.u[1] = ((const uint4*)pb)[1];
        acc = __builtin_amdgcn_wmma_f32_16x16x32_bf16(false, a.v, false, bfr.v, (short)0, acc, false, false);
    }

    // softmax over k-head axis (columns = 16-lane groups), scale = 1/sqrt(D)
    for (int v = 0; v < 8; ++v) {
        float x = acc[v] * INV_NORM;
        float mx = x;
        for (int off = 1; off < 16; off <<= 1) mx = fmaxf(mx, __shfl_xor(mx, off, 16));
        float ex = __expf(x - mx);
        float sm = ex;
        for (int off = 1; off < 16; off <<= 1) sm += __shfl_xor(sm, off, 16);
        attn_lds[wave][sel * 8 + v][r] = ex / sm;
    }
    __syncthreads();

    // ctx[i][d] = sum_j attn[i][j] * v[j][d]; lane owns 4 dims
    const float* vbase = fused + (size_t)t * NQKV + 256;   // v component offset
    const int d0 = lane * 4;
    float c[16][4];
    for (int i = 0; i < 16; ++i)
        for (int qd = 0; qd < 4; ++qd) c[i][qd] = 0.f;
    for (int j = 0; j < 16; ++j) {
        float4 vj = *(const float4*)(vbase + j * 384 + d0);
        for (int i = 0; i < 16; ++i) {
            float aij = attn_lds[wave][i][j];
            c[i][0] += aij * vj.x; c[i][1] += aij * vj.y;
            c[i][2] += aij * vj.z; c[i][3] += aij * vj.w;
        }
    }
    for (int i = 0; i < 16; ++i) {
        union { __bf16 h[4]; unsigned long long u; } o;
        o.h[0] = f2bf(c[i][0]); o.h[1] = f2bf(c[i][1]);
        o.h[2] = f2bf(c[i][2]); o.h[3] = f2bf(c[i][3]);
        *(unsigned long long*)(ctxb + (size_t)t * HID + i * 128 + d0) = o.u;
    }
}

// ---------------------------------------------------------------------------
// Host-side orchestration
// ---------------------------------------------------------------------------
extern "C" void kernel_launch(void* const* d_in, const int* in_sizes, int n_in,
                              void* d_out, int out_size, void* d_ws, size_t ws_size,
                              hipStream_t stream) {
    const float* hidden   = (const float*)d_in[0];
    const float* residual = (const float*)d_in[1];
    const float* alibi    = (const float*)d_in[2];
    // d_in[3] attention_mask: zeros, unused by reference math
    const float* qkv_w    = (const float*)d_in[4];
    const float* qkv_b    = (const float*)d_in[5];
    const float* dense_w  = (const float*)d_in[6];
    const float* dense_b  = (const float*)d_in[7];
    float* out = (float*)d_out;

    char* ws = (char*)d_ws;
    size_t off = 0;
    auto alloc = [&](size_t bytes) -> void* {
        void* p = ws + off;
        off = (off + bytes + 255) & ~(size_t)255;
        return p;
    };
    __bf16* hsb     = (__bf16*)alloc((size_t)TT * HID * 2);
    __bf16* qkvwb   = (__bf16*)alloc((size_t)NQKV * HID * 2);
    __bf16* densewb = (__bf16*)alloc((size_t)HID * HID * 2);
    float*  fused   = (float*)alloc((size_t)TT * NQKV * 4);
    __bf16* qb      = (__bf16*)alloc((size_t)TT * HID * 2);
    __bf16* koffb   = (__bf16*)alloc((size_t)TT * HID * 2);
    float*  Ginv    = (float*)alloc((size_t)BB * NHEADS * HD * HD * 4);
    __bf16* ctxb    = (__bf16*)alloc((size_t)TT * HID * 2);

    // 1) bf16 casts
    long n4;
    n4 = (long)TT * HID / 4;
    cast4_kernel<<<(unsigned)((n4 + 255) / 256), 256, 0, stream>>>((const float4*)hidden, hsb, n4);
    n4 = (long)NQKV * HID / 4;
    cast4_kernel<<<(unsigned)((n4 + 255) / 256), 256, 0, stream>>>((const float4*)qkv_w, qkvwb, n4);
    n4 = (long)HID * HID / 4;
    cast4_kernel<<<(unsigned)((n4 + 255) / 256), 256, 0, stream>>>((const float4*)dense_w, densewb, n4);

    // 2) QKV GEMM (+ q extraction as bf16), async-LDS double-buffered
    wmma_gemm_kernel<<<dim3(NQKV / 128, TT / 64), 256, 0, stream>>>(
        hsb, qkvwb, qkv_b, nullptr, fused, qb, TT, NQKV, HID);

    // 3) pinv path: Gram, invert, k_off
    gram_kernel<<<BB * NHEADS, 256, 0, stream>>>(fused, Ginv);
    invert128_kernel<<<BB * NHEADS, 256, 0, stream>>>(Ginv);
    koff_kernel<<<TT * NHEADS, 128, 0, stream>>>(fused, Ginv, alibi, koffb);

    // 4) per-token head attention (WMMA 16x16 scores + softmax + attn@v)
    attn_kernel<<<TT / 8, 256, 0, stream>>>(qb, koffb, fused, ctxb);

    // 5) dense GEMM + bias + residual -> out
    wmma_gemm_kernel<<<dim3(HID / 128, TT / 64), 256, 0, stream>>>(
        ctxb, densewb, dense_b, residual, out, nullptr, TT, HID, HID);
}
